// GraphConvSparse_44066364457050
// MI455X (gfx1250) — matrix operand order repaired and checked by reading, not compiled, not run on hardware
//
#include <hip/hip_runtime.h>
#include <hip/hip_bf16.h>
#include <stdint.h>

// GraphConvSparse for MI455X (gfx1250, wave32):
//   W_ortho = W @ inv(chol(W^T W + 1e-4 I))^T        (tiny, 3 kernels)
//   X       = inputs @ W_ortho                        (bf16 WMMA, f32 accum;
//                                                      W staged via TDM when available)
//   out     = tanh(segment_sum(vals * X[cols], rows)) (CSR build + gather SpMM)

#define D_IN  512
#define D_OUT 128

typedef __attribute__((ext_vector_type(16))) __bf16 v16bf;
typedef __attribute__((ext_vector_type(8)))  float  v8f;
typedef unsigned int u32x4 __attribute__((ext_vector_type(4)));
typedef int          i32x4 __attribute__((ext_vector_type(4)));
typedef int          i32x8 __attribute__((ext_vector_type(8)));

#if defined(__AMDGCN__) && __has_builtin(__builtin_amdgcn_tensor_load_to_lds) && \
    __has_builtin(__builtin_amdgcn_s_wait_tensorcnt)
#define USE_TDM 1
#if __has_include(<hip/amd_detail/amd_gfx1250_TDM.h>)
#define TDM_6ARG 1   // amdgpu-toolchain (therock-10.0 headers) => 6-arg builtin
#endif
#endif

__device__ __forceinline__ uint32_t bf16pk(float a, float b) {
  // round-to-nearest-even f32 -> bf16, packed pair
  uint32_t ua = __float_as_uint(a), ub = __float_as_uint(b);
  uint32_t ra = (ua + 0x7FFFu + ((ua >> 16) & 1u)) >> 16;
  uint32_t rb = (ub + 0x7FFFu + ((ub >> 16) & 1u)) >> 16;
  return ra | (rb << 16);
}

// ---------------- ortho-norm: G = W^T W + 1e-4 I ----------------
__global__ void __launch_bounds__(256) k_gram(const float* __restrict__ W,
                                              float* __restrict__ G) {
  int t = threadIdx.x;
  int bi = blockIdx.x >> 3, bj = blockIdx.x & 7;
  int i = bi * 16 + (t >> 4);
  int j = bj * 16 + (t & 15);
  float s = 0.f;
  for (int k = 0; k < D_IN; ++k)
    s = fmaf(W[k * D_OUT + i], W[k * D_OUT + j], s);
  G[i * D_OUT + j] = s + (i == j ? 1e-4f : 0.f);
}

// ---------------- in-LDS Cholesky of 128x128 ----------------
__global__ void __launch_bounds__(128) k_chol(const float* __restrict__ G,
                                              float* __restrict__ L) {
  __shared__ float s[D_OUT * D_OUT];   // 64 KB
  int t = threadIdx.x;
  for (int idx = t; idx < D_OUT * D_OUT; idx += 128) s[idx] = G[idx];
  __syncthreads();
  for (int k = 0; k < D_OUT; ++k) {
    if (t == 0) s[k * D_OUT + k] = sqrtf(s[k * D_OUT + k]);
    __syncthreads();
    float inv = 1.0f / s[k * D_OUT + k];
    if (t > k) s[t * D_OUT + k] *= inv;
    __syncthreads();
    if (t > k) {
      float lik = s[t * D_OUT + k];
      for (int j = k + 1; j <= t; ++j)
        s[t * D_OUT + j] = fmaf(-lik, s[j * D_OUT + k], s[t * D_OUT + j]);
    }
    __syncthreads();
  }
  for (int idx = t; idx < D_OUT * D_OUT; idx += 128) {
    int r = idx / D_OUT, c = idx % D_OUT;
    L[idx] = (r >= c) ? s[idx] : 0.f;
  }
}

// ------- forward substitution: solve L Y = W^T; emit W_ortho as bf16 -------
// Output pre-swizzled into a WMMA B-fragment order with contiguous K per lane half (same as A):
//   frag[kt(0..15)][nt(0..7)][lane(0..31)][j(0..15)]
//   lane = 16*(K&31>=16) + (N&15),  j = K&15,  value = W_ortho[K][N]
__global__ void __launch_bounds__(128) k_trisolve(const float* __restrict__ W,
                                                  const float* __restrict__ L,
                                                  uint16_t* __restrict__ fragW) {
  __shared__ float sL[D_OUT * D_OUT];  // 64 KB, broadcast reads
  int t = threadIdx.x;
  for (int idx = t; idx < D_OUT * D_OUT; idx += 128) sL[idx] = L[idx];
  __syncthreads();
  int c = blockIdx.x * 128 + t;        // K-row of W_ortho, 0..511
  int kt = c >> 5, kk = c & 31, khalf = kk >> 4, j = kk & 15;
  float y[D_OUT];                      // per-thread, spills to scratch (ok)
  for (int i = 0; i < D_OUT; ++i) {
    float acc = W[c * D_OUT + i];      // W^T[i][c]
    for (int k = 0; k < i; ++k) acc = fmaf(-sL[i * D_OUT + k], y[k], acc);
    float yi = acc / sL[i * D_OUT + i];
    y[i] = yi;
    int nt = i >> 4, lane = khalf * 16 + (i & 15);
    uint32_t u = __float_as_uint(yi);
    fragW[(((kt * 8 + nt) * 32) + lane) * 16 + j] =
        (uint16_t)((u + 0x7FFFu + ((u >> 16) & 1u)) >> 16);
  }
}

// ---------------- CSR build ----------------
__global__ void k_zero(uint32_t* p, int n) {
  int i = blockIdx.x * blockDim.x + threadIdx.x;
  if (i < n) p[i] = 0u;
}

__global__ void k_count(const int* __restrict__ rows, uint32_t* cnt, int nE) {
  int e = blockIdx.x * blockDim.x + threadIdx.x;
  if (e < nE) atomicAdd(&cnt[rows[e]], 1u);
}

__global__ void __launch_bounds__(1024) k_scan(const uint32_t* __restrict__ cnt,
                                               uint32_t* __restrict__ rowptr,
                                               uint32_t* __restrict__ cursor,
                                               int n) {
  __shared__ uint32_t s[1024];
  __shared__ uint32_t carry;
  int t = threadIdx.x;
  if (t == 0) carry = 0u;
  __syncthreads();
  int nchunks = (n + 1023) / 1024;
  for (int ch = 0; ch < nchunks; ++ch) {
    int i = ch * 1024 + t;
    uint32_t v = (i < n) ? cnt[i] : 0u;
    s[t] = v;
    __syncthreads();
    for (int off = 1; off < 1024; off <<= 1) {  // Hillis-Steele inclusive scan
      uint32_t add = (t >= off) ? s[t - off] : 0u;
      __syncthreads();
      s[t] += add;
      __syncthreads();
    }
    uint32_t excl = carry + s[t] - v;
    if (i < n) { rowptr[i] = excl; cursor[i] = excl; }
    __syncthreads();
    if (t == 1023) carry += s[1023];
    __syncthreads();
  }
  if (t == 0) rowptr[n] = carry;
}

__global__ void k_fill(const int* __restrict__ rows, const int* __restrict__ cols,
                       const float* __restrict__ vals, uint32_t* cursor,
                       int* __restrict__ ccol, float* __restrict__ cval, int nE) {
  int e = blockIdx.x * blockDim.x + threadIdx.x;
  if (e < nE) {
    uint32_t p = atomicAdd(&cursor[rows[e]], 1u);
    ccol[p] = cols[e];
    cval[p] = vals[e];
  }
}

// ---------------- X = inputs @ W_ortho via bf16 WMMA ----------------
// 256 threads = 8 waves; tile = 128 rows x 128 cols; K staged in chunks of 64.
// W chunk (16 KB, contiguous, fragment-ordered) staged by the Tensor Data
// Mover when the builtin is available; A chunk staged manually (needs f32->bf16).
__global__ void __launch_bounds__(256) k_gemm(const float* __restrict__ Xin,
                                              const uint16_t* __restrict__ fragW,
                                              float* __restrict__ Xout,
                                              int nNodes) {
  __shared__ uint32_t sW[4096];      // 2kt x 8nt x 32 lanes x 16 bf16 = 16 KB
  __shared__ uint32_t sA[128 * 36];  // 128 rows x 64 K, bf16, pitch 72 (pad 8)
  const int t = threadIdx.x;
  const int lane = t & 31, w = t >> 5;
  const int mrow = lane & 15;
  const int hi = lane >> 4;
  const int rowBlk = blockIdx.x * 128;

  v8f acc[8] = {};

  for (int cc = 0; cc < 8; ++cc) {
    // ---- stage A chunk: rows rowBlk..+127, K = cc*64..+63, f32 -> bf16 ----
    #pragma unroll
    for (int it = 0; it < 8; ++it) {
      int idx = (it * 256 + t) * 4;            // flat elem idx in 128x64
      int r = idx >> 6, k = idx & 63;
      int grow = rowBlk + r;
      float4 f = make_float4(0.f, 0.f, 0.f, 0.f);
      if (grow < nNodes)
        f = *reinterpret_cast<const float4*>(
            &Xin[(size_t)grow * D_IN + cc * 64 + k]);
      int dw = r * 36 + (k >> 1);
      sA[dw]     = bf16pk(f.x, f.y);
      sA[dw + 1] = bf16pk(f.z, f.w);
    }
    // ---- stage W chunk: 16 KB contiguous global -> LDS ----
#ifdef USE_TDM
    if (w == 0) {
      // Tensor DMA descriptor (ISA 8.3/8.4): 1-D tile, 2048 elements of 8 B.
      uint64_t ga = (uint64_t)(uintptr_t)fragW + (uint64_t)cc * 16384u;
      uint32_t lds = (uint32_t)(uintptr_t)(&sW[0]);
      u32x4 g0 = { 1u,                                   // count=1 (valid)
                   lds,                                  // lds_addr (bytes)
                   (uint32_t)ga,                         // global_addr[31:0]
                   (uint32_t)((ga >> 32) & 0x01FFFFFFull) | (2u << 30) }; // type=2
      i32x8 g1 = { (int)(3u << 16),        // workgroup_mask=0, data_size=3 (8 B)
                   (int)(2048u << 16),     // tensor_dim0[15:0]=2048  (bits 79:48)
                   (int)(1u << 16),        // tensor_dim0 hi=0 | tensor_dim1=1
                   (int)(2048u << 16),     // tensor_dim1 hi=0 | tile_dim0=2048
                   1,                      // tile_dim1=1, tile_dim2=0
                   2048,                   // tensor_dim0_stride[31:0]
                   0, 0 };
      i32x4 gz = { 0, 0, 0, 0 };
#ifdef TDM_6ARG
      i32x8 gz8 = { 0, 0, 0, 0, 0, 0, 0, 0 };
      __builtin_amdgcn_tensor_load_to_lds(g0, g1, gz, gz, gz8, 0);
#else
      __builtin_amdgcn_tensor_load_to_lds(g0, g1, gz, gz, 0);
#endif
      __builtin_amdgcn_s_wait_tensorcnt(0);  // done before barrier signal
    }
#else
    {
      const uint4* g = reinterpret_cast<const uint4*>(fragW + (size_t)cc * 8192);
      uint4* s4 = reinterpret_cast<uint4*>(sW);
      #pragma unroll
      for (int it = 0; it < 4; ++it) s4[it * 256 + t] = g[it * 256 + t];
    }
#endif
    __syncthreads();

    #pragma unroll
    for (int kt2 = 0; kt2 < 2; ++kt2) {
      // A fragment: lanes 0-15 K{0..15}, lanes 16-31 K{16..31}
      int kbase = kt2 * 32 + hi * 16;
      const uint4* pa = reinterpret_cast<const uint4*>(
          reinterpret_cast<const char*>(sA) + ((w * 16 + mrow) * 144 + kbase * 2));
      union BU { uint4 u[2]; v16bf v; };
      BU A;
      A.u[0] = pa[0];
      A.u[1] = pa[1];  // +16 bytes: K base + 8
      // hoist all 8 B fragments so waits amortize across the WMMA chain
      BU B[8];
      #pragma unroll
      for (int nt = 0; nt < 8; ++nt) {
        const uint4* pb = reinterpret_cast<const uint4*>(sW) +
                          ((kt2 * 8 + nt) * 32 + lane) * 2;
        B[nt].u[0] = pb[0];
        B[nt].u[1] = pb[1];
      }
      #pragma unroll
      for (int nt = 0; nt < 8; ++nt)
        acc[nt] = __builtin_amdgcn_wmma_f32_16x16x32_bf16(
            false, A.v, false, B[nt].v, (short)0, acc[nt], false, false);
    }
    __syncthreads();
  }

  // C layout: lane -> N = lane&15; vgpr v -> M = v + 8*(lane>=16)
  #pragma unroll
  for (int nt = 0; nt < 8; ++nt) {
    int col = nt * 16 + mrow;
    #pragma unroll
    for (int v = 0; v < 8; ++v) {
      int row = rowBlk + w * 16 + (hi ? v + 8 : v);
      if (row < nNodes) Xout[(size_t)row * D_OUT + col] = acc[nt][v];
    }
  }
}

// ------------- CSR SpMM + tanh (one wave per destination row) -------------
__global__ void __launch_bounds__(256) k_spmm(const float* __restrict__ X,
                                              const uint32_t* __restrict__ rowptr,
                                              const int* __restrict__ ccol,
                                              const float* __restrict__ cval,
                                              float* __restrict__ out,
                                              int nNodes) {
  int lane = threadIdx.x & 31;
  int w = threadIdx.x >> 5;
  int row = blockIdx.x * 8 + w;
  if (row >= nNodes) return;
  uint32_t s0 = rowptr[row], s1 = rowptr[row + 1];
  float4 acc = make_float4(0.f, 0.f, 0.f, 0.f);
  for (uint32_t base = s0; base < s1; base += 32u) {
    uint32_t e = base + (uint32_t)lane;
    int c = 0; float v = 0.f;
    if (e < s1) {
      c = ccol[e];
      v = cval[e];
      __builtin_prefetch(&X[(size_t)c * D_OUT + lane * 4], 0, 0);  // global_prefetch_b8
    }
    int m = (int)min(32u, s1 - base);
    for (int jj = 0; jj < m; ++jj) {
      int   src = __shfl(c, jj, 32);
      float vv  = __shfl(v, jj, 32);
      float4 x = *reinterpret_cast<const float4*>(&X[(size_t)src * D_OUT + lane * 4]);
      acc.x = fmaf(vv, x.x, acc.x);
      acc.y = fmaf(vv, x.y, acc.y);
      acc.z = fmaf(vv, x.z, acc.z);
      acc.w = fmaf(vv, x.w, acc.w);
    }
  }
  float4 o;
  o.x = tanhf(acc.x); o.y = tanhf(acc.y); o.z = tanhf(acc.z); o.w = tanhf(acc.w);
  *reinterpret_cast<float4*>(&out[(size_t)row * D_OUT + lane * 4]) = o;
}

// ---------------- launch ----------------
extern "C" void kernel_launch(void* const* d_in, const int* in_sizes, int n_in,
                              void* d_out, int out_size, void* d_ws, size_t ws_size,
                              hipStream_t stream) {
  const float* inputs = (const float*)d_in[0];
  const float* weight = (const float*)d_in[1];
  const float* vals   = (const float*)d_in[2];
  const int*   rows   = (const int*)d_in[3];
  const int*   cols   = (const int*)d_in[4];
  const int nNodes = in_sizes[0] / D_IN;
  const int nEdges = in_sizes[2];

  char* ws = (char*)d_ws;
  size_t off = 0;
  auto alloc = [&](size_t bytes) -> char* {
    char* p = ws + off;
    off += (bytes + 255) & ~(size_t)255;
    return p;
  };
  float*    G      = (float*)alloc((size_t)D_OUT * D_OUT * 4);
  float*    L      = (float*)alloc((size_t)D_OUT * D_OUT * 4);
  uint16_t* fragW  = (uint16_t*)alloc((size_t)D_IN * D_OUT * 2);
  float*    X      = (float*)alloc((size_t)nNodes * D_OUT * 4);
  uint32_t* cnt    = (uint32_t*)alloc((size_t)nNodes * 4);
  uint32_t* rowptr = (uint32_t*)alloc(((size_t)nNodes + 1) * 4);
  uint32_t* cursor = (uint32_t*)alloc((size_t)nNodes * 4);
  int*      ccol   = (int*)alloc((size_t)nEdges * 4);
  float*    cval   = (float*)alloc((size_t)nEdges * 4);
  (void)ws_size; (void)n_in; (void)out_size;

  k_zero<<<(nNodes + 255) / 256, 256, 0, stream>>>(cnt, nNodes);
  k_gram<<<64, 256, 0, stream>>>(weight, G);
  k_chol<<<1, 128, 0, stream>>>(G, L);
  k_trisolve<<<4, 128, 0, stream>>>(weight, L, fragW);
  k_count<<<(nEdges + 255) / 256, 256, 0, stream>>>(rows, cnt, nEdges);
  k_scan<<<1, 1024, 0, stream>>>(cnt, rowptr, cursor, nNodes);
  k_fill<<<(nEdges + 255) / 256, 256, 0, stream>>>(rows, cols, vals, cursor,
                                                   ccol, cval, nEdges);
  k_gemm<<<(nNodes + 127) / 128, 256, 0, stream>>>(inputs, fragW, X, nNodes);
  k_spmm<<<(nNodes + 7) / 8, 256, 0, stream>>>(X, rowptr, ccol, cval,
                                               (float*)d_out, nNodes);
}
